// MultiHeadAttentionParallel_16870631538975
// MI455X (gfx1250) — compile-verified, matrix operand order checked by reading, not run on hardware
//
#include <hip/hip_runtime.h>
#include <hip/hip_bf16.h>

typedef unsigned short u16;
typedef u16   v8u    __attribute__((ext_vector_type(8)));
typedef u16   v16u   __attribute__((ext_vector_type(16)));
typedef __bf16 bf16x16 __attribute__((ext_vector_type(16)));
typedef float v8f    __attribute__((ext_vector_type(8)));

#define SCALE_ 0.125f   // D^-0.5, D=64

// ---------- helpers ----------
__device__ __forceinline__ u16 f2bf(float f) {
  union { float f; unsigned u; } v; v.f = f;
  unsigned r = v.u + 0x7FFFu + ((v.u >> 16) & 1u);   // round-to-nearest-even
  return (u16)(r >> 16);
}
__device__ __forceinline__ float bf2f(u16 b) {
  union { unsigned u; float f; } v; v.u = ((unsigned)b) << 16;
  return v.f;
}
__device__ __forceinline__ v8f vzero8() {
  v8f z;
#pragma unroll
  for (int i = 0; i < 8; ++i) z[i] = 0.0f;
  return z;
}

// ---- DPP 16-lane butterfly reductions (no LDS, no dscnt waits) ----
// stages: quad_perm xor1 (0xB1), quad_perm xor2 (0x4E), row_half_mirror (0x141),
// row_mirror (0x140). After each stage sub-groups are uniform, so mirrors act as
// xor4/xor8. All ctrls stay within a 16-lane row (wave32 = 2 rows).
template <int CTRL>
__device__ __forceinline__ float dppf(float x) {
  return __builtin_bit_cast(float,
      __builtin_amdgcn_update_dpp(0, __builtin_bit_cast(int, x), CTRL, 0xF, 0xF, true));
}
__device__ __forceinline__ float red16_max(float x) {
  x = fmaxf(x, dppf<0xB1>(x));
  x = fmaxf(x, dppf<0x4E>(x));
  x = fmaxf(x, dppf<0x141>(x));
  x = fmaxf(x, dppf<0x140>(x));
  return x;
}
__device__ __forceinline__ float red16_add(float x) {
  x += dppf<0xB1>(x);
  x += dppf<0x4E>(x);
  x += dppf<0x141>(x);
  x += dppf<0x140>(x);
  return x;
}

// ---- async global -> LDS (CDNA5 TDM-class path, tracked by ASYNCcnt) ----
// GVS mode: mem = SGPR base + per-lane i32 byte offset. LDS byte address is the
// low 32 bits of the generic pointer.
__device__ __forceinline__ void async_b128(const void* gbase, unsigned goff, void* ldsaddr) {
  asm volatile("global_load_async_to_lds_b128 %0, %1, %2"
               :: "v"((unsigned)(size_t)ldsaddr), "v"(goff), "s"(gbase)
               : "memory");
}
#define ASYNC_WAIT_ALL()  asm volatile("s_wait_asynccnt 0x0" ::: "memory")
#define ASYNC_WAIT_3()    asm volatile("s_wait_asynccnt 0x3" ::: "memory")

// WMMA bf16 16x16x32 fragment loader (identical for A row-major and B from B^T rows).
__device__ __forceinline__ bf16x16 ldfrag(const u16* rowbase, int hf) {
  v8u lo = *(const v8u*)(rowbase + hf * 8);
  v8u hi = *(const v8u*)(rowbase + 16 + hf * 8);
  v16u r;
#pragma unroll
  for (int i = 0; i < 8; ++i) { r[i] = lo[i]; r[8 + i] = hi[i]; }
  return __builtin_bit_cast(bf16x16, r);
}
#define WMMA_BF16(A, Bm, C) \
  __builtin_amdgcn_wmma_f32_16x16x32_bf16(false, (A), false, (Bm), (short)0, (C), false, false)

// ---------- prep: W[in][out] (f32) -> Wt[out][in] (bf16) ----------
__global__ void transpose_to_bf16(const float* __restrict__ W, u16* __restrict__ Wt) {
  int idx = blockIdx.x * 256 + threadIdx.x;   // 2304*256 == 768*768
  int o = idx / 768, e = idx % 768;
  Wt[o * 768 + e] = f2bf(W[e * 768 + o]);
}

// rates[0..63]=softplus(ak), rates[64..127]=softplus(av)
__global__ void prep_rates(const float* __restrict__ ak, const float* __restrict__ av,
                           float* __restrict__ rates) {
  int t = threadIdx.x;            // 128 threads
  float a = (t < 64) ? ak[t] : av[t - 64];
  rates[t] = log1pf(__expf(a));
}

// ---------- LayerNorm: x[9216][768] f32 -> xn bf16 ----------
__global__ void __launch_bounds__(256) layernorm_kernel(const float* __restrict__ x,
                                                        const float* __restrict__ gw,
                                                        const float* __restrict__ bw,
                                                        u16* __restrict__ xn) {
  __shared__ float red[8];
  __shared__ float s_mu, s_rs;
  const int row = blockIdx.x, tid = threadIdx.x;
  const float* xr = x + (size_t)row * 768;
  float v0 = xr[tid], v1 = xr[tid + 256], v2 = xr[tid + 512];
  float s = v0 + v1 + v2;
#pragma unroll
  for (int m = 16; m >= 1; m >>= 1) s += __shfl_xor(s, m, 32);
  if ((tid & 31) == 0) red[tid >> 5] = s;
  __syncthreads();
  if (tid == 0) {
    float t = 0.f;
#pragma unroll
    for (int i = 0; i < 8; ++i) t += red[i];
    s_mu = t * (1.0f / 768.0f);
  }
  __syncthreads();
  float mu = s_mu;
  float d0 = v0 - mu, d1 = v1 - mu, d2 = v2 - mu;
  float vs = d0 * d0 + d1 * d1 + d2 * d2;
#pragma unroll
  for (int m = 16; m >= 1; m >>= 1) vs += __shfl_xor(vs, m, 32);
  if ((tid & 31) == 0) red[tid >> 5] = vs;
  __syncthreads();
  if (tid == 0) {
    float t = 0.f;
#pragma unroll
    for (int i = 0; i < 8; ++i) t += red[i];
    s_rs = rsqrtf(t * (1.0f / 768.0f) + 1e-5f);
  }
  __syncthreads();
  float rs = s_rs;
  u16* xo = xn + (size_t)row * 768;
  xo[tid]       = f2bf(d0 * rs * gw[tid]       + bw[tid]);
  xo[tid + 256] = f2bf(d1 * rs * gw[tid + 256] + bw[tid + 256]);
  xo[tid + 512] = f2bf(d2 * rs * gw[tid + 512] + bw[tid + 512]);
}

// ---------- GEMM: C[M=9216][N=768] = A[M][768] * Bt[N][768]^T + bias ----------
// workgroup tile 128x64, 8 waves in 4x2 grid, each wave 32x32 (2x2 WMMA tiles).
// Double-buffered async global->LDS DMA overlapped with WMMA.
template <bool BF16OUT>
__global__ void __launch_bounds__(256) gemm_bias_kernel(const u16* __restrict__ A,
                                                        const u16* __restrict__ Bt,
                                                        const float* __restrict__ bias,
                                                        void* __restrict__ Cout) {
  __shared__ __align__(16) u16 As[2][128 * 40];   // stride 40 elems (80B, 16B aligned)
  __shared__ __align__(16) u16 Bs[2][64 * 40];
  const int tid = threadIdx.x;
  const int wv = tid >> 5, lane = tid & 31;
  const int hf = lane >> 4, line = lane & 15;
  const int wm = wv >> 1, wn = wv & 1;
  const int m0 = blockIdx.x * 128, n0 = blockIdx.y * 64;

  v8f acc[2][2] = { { vzero8(), vzero8() }, { vzero8(), vzero8() } };

  const int arow = tid >> 1, acb = (tid & 1) * 16;
  const int brow = tid >> 2, bcb = (tid & 3) * 8;
  const unsigned abase = ((unsigned)(m0 + arow) * 768 + (unsigned)acb) * 2u;
  const unsigned bbase = ((unsigned)(n0 + brow) * 768 + (unsigned)bcb) * 2u;

  // prologue: fetch first tile into buffer 0
  async_b128(A, abase, &As[0][arow * 40 + acb]);
  async_b128(A, abase + 16u, &As[0][arow * 40 + acb + 8]);
  async_b128(Bt, bbase, &Bs[0][brow * 40 + bcb]);

  int buf = 0;
  for (int kk = 0; kk < 768; kk += 32) {
    const bool hasnext = (kk + 32) < 768;
    if (hasnext) {   // issue next tile into the other buffer (its readers passed last barrier)
      unsigned ko = (unsigned)(kk + 32) * 2u;
      async_b128(A, abase + ko, &As[buf ^ 1][arow * 40 + acb]);
      async_b128(A, abase + ko + 16u, &As[buf ^ 1][arow * 40 + acb + 8]);
      async_b128(Bt, bbase + ko, &Bs[buf ^ 1][brow * 40 + bcb]);
      ASYNC_WAIT_3();        // async loads complete in order: current buffer is done
    } else {
      ASYNC_WAIT_ALL();
    }
    __syncthreads();
    const u16* as = As[buf];
    const u16* bs = Bs[buf];
    bf16x16 fa0 = ldfrag(as + (wm * 32 + line) * 40, hf);
    bf16x16 fa1 = ldfrag(as + (wm * 32 + 16 + line) * 40, hf);
    bf16x16 fb0 = ldfrag(bs + (wn * 32 + line) * 40, hf);
    bf16x16 fb1 = ldfrag(bs + (wn * 32 + 16 + line) * 40, hf);
    acc[0][0] = WMMA_BF16(fa0, fb0, acc[0][0]);
    acc[0][1] = WMMA_BF16(fa0, fb1, acc[0][1]);
    acc[1][0] = WMMA_BF16(fa1, fb0, acc[1][0]);
    acc[1][1] = WMMA_BF16(fa1, fb1, acc[1][1]);
    __syncthreads();
    buf ^= 1;
  }
#pragma unroll
  for (int ti = 0; ti < 2; ++ti) {
#pragma unroll
    for (int tj = 0; tj < 2; ++tj) {
      int col = n0 + wn * 32 + tj * 16 + line;
      float bvv = bias[col];
#pragma unroll
      for (int j = 0; j < 8; ++j) {
        int row = m0 + wm * 32 + ti * 16 + hf * 8 + j;
        float val = acc[ti][tj][j] + bvv;
        if (BF16OUT) ((u16*)Cout)[(size_t)row * 768 + col] = f2bf(val);
        else        ((float*)Cout)[(size_t)row * 768 + col] = val;
      }
    }
  }
}

// ---------- fused attention ----------
// grid (s=3, c=3, g=192); WG handles 64 query rows r0..r0+63 vs all 576 keys.
// Keff = K + exp(-dist*rateK), Veff = V + exp(-dist*rateV) built per 64-key chunk in LDS.
// Two passes: (1) row max via WMMA + DPP reductions, (2) exp/sum + P@Veff accumulation.
__global__ void __launch_bounds__(256) attention_kernel(const u16* __restrict__ Q,
                                                        const u16* __restrict__ K,
                                                        const u16* __restrict__ V,
                                                        const float* __restrict__ dist,
                                                        const float* __restrict__ rates,
                                                        u16* __restrict__ O) {
  __shared__ __align__(16) u16 Qs[64 * 72];    // stride 72 elems (144B, 16B aligned)
  __shared__ __align__(16) u16 Ks[64 * 72];    // Keff[m][d]
  __shared__ __align__(16) u16 Vts[64 * 72];   // Veff^T[d][m]
  __shared__ __align__(16) u16 Ps[64 * 72];    // P[row][m]
  __shared__ float partial[4][64];
  __shared__ float rowmax[64];
  __shared__ float rowsum[64];
  __shared__ float rateK[64];
  __shared__ float rateV[64];

  const int s = blockIdx.x, c = blockIdx.y, g = blockIdx.z;
  const int r0 = g * 576 + c * 192 + s * 64;
  const int rp = g * 3 + c;
  const int tid = threadIdx.x;
  const int wv = tid >> 5, lane = tid & 31;
  const int hf = lane >> 4, line = lane & 15;
  const int ti0 = (wv * 2) >> 2,     tj0 = (wv * 2) & 3;
  const int ti1 = (wv * 2 + 1) >> 2, tj1 = (wv * 2 + 1) & 3;

  if (tid < 64) {
    rowmax[tid] = -3.0e38f;
    rowsum[tid] = 0.0f;
    rateK[tid] = rates[tid];
    rateV[tid] = rates[64 + tid];
  }
  {   // async DMA the 64x64 Q block into LDS
    int row = tid >> 2, cb = (tid & 3) * 16;
    unsigned goff = ((unsigned)(r0 + row) * 64 + (unsigned)cb) * 2u;
    async_b128(Q, goff, Qs + row * 72 + cb);
    async_b128(Q, goff + 16u, Qs + row * 72 + cb + 8);
    ASYNC_WAIT_ALL();
  }
  __syncthreads();

  const int em = tid >> 2;          // local key index 0..63
  const int ed0 = (tid & 3) * 16;   // d base
  v8f oacc0 = vzero8(), oacc1 = vzero8();

  // ---- pass 1: row max ----
  for (int mc = 0; mc < 9; ++mc) {
    const int m0 = mc * 64;
    {
      float dv = dist[(size_t)rp * 576 + m0 + em];
      const u16* krow = K + (size_t)(g * 576 + m0 + em) * 64;
      if (mc + 1 < 9) __builtin_prefetch(krow + 64 * 64 + ed0, 0, 1);
#pragma unroll
      for (int jj = 0; jj < 16; ++jj) {
        int d = ed0 + jj;
        Ks[em * 72 + d] = f2bf(bf2f(krow[d]) + __expf(-dv * rateK[d]));
      }
    }
    __syncthreads();
#pragma unroll
    for (int it = 0; it < 2; ++it) {
      const int ti = it ? ti1 : ti0, tj = it ? tj1 : tj0;
      bf16x16 a0 = ldfrag(Qs + (ti * 16 + line) * 72, hf);
      bf16x16 a1 = ldfrag(Qs + (ti * 16 + line) * 72 + 32, hf);
      bf16x16 b0 = ldfrag(Ks + (tj * 16 + line) * 72, hf);
      bf16x16 b1 = ldfrag(Ks + (tj * 16 + line) * 72 + 32, hf);
      v8f sa = vzero8();
      sa = WMMA_BF16(a0, b0, sa);
      sa = WMMA_BF16(a1, b1, sa);
#pragma unroll
      for (int j = 0; j < 8; ++j) {
        float v = red16_max(sa[j]);
        if (line == 0) partial[tj][ti * 16 + hf * 8 + j] = v;
      }
    }
    __syncthreads();
    if (tid < 64) {
      float m = fmaxf(fmaxf(partial[0][tid], partial[1][tid]),
                      fmaxf(partial[2][tid], partial[3][tid]));
      rowmax[tid] = fmaxf(rowmax[tid], m);
    }
    __syncthreads();
  }

  // ---- pass 2: exp, row sums, P @ Veff ----
  for (int mc = 0; mc < 9; ++mc) {
    const int m0 = mc * 64;
    {
      float dv = dist[(size_t)rp * 576 + m0 + em];
      const u16* krow = K + (size_t)(g * 576 + m0 + em) * 64;
      const u16* vrow = V + (size_t)(g * 576 + m0 + em) * 64;
      if (mc + 1 < 9) {
        __builtin_prefetch(krow + 64 * 64 + ed0, 0, 1);
        __builtin_prefetch(vrow + 64 * 64 + ed0, 0, 1);
      }
#pragma unroll
      for (int jj = 0; jj < 16; ++jj) {
        int d = ed0 + jj;
        Ks[em * 72 + d]  = f2bf(bf2f(krow[d]) + __expf(-dv * rateK[d]));
        Vts[d * 72 + em] = f2bf(bf2f(vrow[d]) + __expf(-dv * rateV[d]));
      }
    }
    __syncthreads();
#pragma unroll
    for (int it = 0; it < 2; ++it) {
      const int ti = it ? ti1 : ti0, tj = it ? tj1 : tj0;
      bf16x16 a0 = ldfrag(Qs + (ti * 16 + line) * 72, hf);
      bf16x16 a1 = ldfrag(Qs + (ti * 16 + line) * 72 + 32, hf);
      bf16x16 b0 = ldfrag(Ks + (tj * 16 + line) * 72, hf);
      bf16x16 b1 = ldfrag(Ks + (tj * 16 + line) * 72 + 32, hf);
      v8f sa = vzero8();
      sa = WMMA_BF16(a0, b0, sa);
      sa = WMMA_BF16(a1, b1, sa);
#pragma unroll
      for (int j = 0; j < 8; ++j) {
        int row = ti * 16 + hf * 8 + j;
        float p = __expf((sa[j] - rowmax[row]) * SCALE_);
        Ps[row * 72 + tj * 16 + line] = f2bf(p);
        float ssum = red16_add(p);
        if (line == 0) partial[tj][row] = ssum;
      }
    }
    __syncthreads();
    if (tid < 64)
      rowsum[tid] += partial[0][tid] + partial[1][tid] + partial[2][tid] + partial[3][tid];
#pragma unroll
    for (int it = 0; it < 2; ++it) {
      const int ti = it ? ti1 : ti0, tjd = it ? tj1 : tj0;
      bf16x16 pa0 = ldfrag(Ps + (ti * 16 + line) * 72, hf);
      bf16x16 pa1 = ldfrag(Ps + (ti * 16 + line) * 72 + 32, hf);
      bf16x16 vb0 = ldfrag(Vts + (tjd * 16 + line) * 72, hf);
      bf16x16 vb1 = ldfrag(Vts + (tjd * 16 + line) * 72 + 32, hf);
      if (it == 0) { oacc0 = WMMA_BF16(pa0, vb0, oacc0); oacc0 = WMMA_BF16(pa1, vb1, oacc0); }
      else         { oacc1 = WMMA_BF16(pa0, vb0, oacc1); oacc1 = WMMA_BF16(pa1, vb1, oacc1); }
    }
    __syncthreads();
  }

  // ---- epilogue: normalize and store (flat [row][d] == faithful raw reshape) ----
#pragma unroll
  for (int it = 0; it < 2; ++it) {
    const int ti = it ? ti1 : ti0, tjd = it ? tj1 : tj0;
#pragma unroll
    for (int j = 0; j < 8; ++j) {
      int row = ti * 16 + hf * 8 + j;
      float inv = 1.0f / rowsum[row];
      float val = (it ? oacc1[j] : oacc0[j]) * inv;
      O[(size_t)(r0 + row) * 64 + tjd * 16 + line] = f2bf(val);
    }
  }
}

// ---------- host ----------
extern "C" void kernel_launch(void* const* d_in, const int* in_sizes, int n_in,
                              void* d_out, int out_size, void* d_ws, size_t ws_size,
                              hipStream_t stream) {
  (void)in_sizes; (void)n_in; (void)out_size; (void)ws_size;
  const float* x    = (const float*)d_in[0];
  const float* dist = (const float*)d_in[1];
  const float* ln_g = (const float*)d_in[2];
  const float* ln_b = (const float*)d_in[3];
  const float* Wq   = (const float*)d_in[4];
  const float* bq   = (const float*)d_in[5];
  const float* Wk   = (const float*)d_in[6];
  const float* bk   = (const float*)d_in[7];
  const float* Wv   = (const float*)d_in[8];
  const float* bv   = (const float*)d_in[9];
  const float* Wo   = (const float*)d_in[10];
  const float* bo   = (const float*)d_in[11];
  const float* ak   = (const float*)d_in[12];
  const float* av   = (const float*)d_in[13];
  float* out = (float*)d_out;

  char* ws = (char*)d_ws;
  size_t off = 0;
  auto alloc = [&](size_t bytes) -> void* {
    void* p = ws + off;
    off = (off + bytes + 255) & ~(size_t)255;
    return p;
  };
  const size_t WTB  = (size_t)768 * 768 * 2;    // bf16 weight matrix
  const size_t PRJB = (size_t)9216 * 768 * 2;   // bf16 activation matrix
  u16* Wtq   = (u16*)alloc(WTB);
  u16* Wtk   = (u16*)alloc(WTB);
  u16* Wtv   = (u16*)alloc(WTB);
  u16* Wto   = (u16*)alloc(WTB);
  float* rts = (float*)alloc(128 * sizeof(float));
  u16* xn    = (u16*)alloc(PRJB);
  u16* qb    = (u16*)alloc(PRJB);
  u16* kb    = (u16*)alloc(PRJB);
  u16* vb    = (u16*)alloc(PRJB);
  u16* attnO = (u16*)alloc(PRJB);

  transpose_to_bf16<<<2304, 256, 0, stream>>>(Wq, Wtq);
  transpose_to_bf16<<<2304, 256, 0, stream>>>(Wk, Wtk);
  transpose_to_bf16<<<2304, 256, 0, stream>>>(Wv, Wtv);
  transpose_to_bf16<<<2304, 256, 0, stream>>>(Wo, Wto);
  prep_rates<<<1, 128, 0, stream>>>(ak, av, rts);

  layernorm_kernel<<<9216, 256, 0, stream>>>(x, ln_g, ln_b, xn);

  dim3 ggrid(72, 12);
  gemm_bias_kernel<true><<<ggrid, 256, 0, stream>>>(xn, Wtq, bq, qb);
  gemm_bias_kernel<true><<<ggrid, 256, 0, stream>>>(xn, Wtk, bk, kb);
  gemm_bias_kernel<true><<<ggrid, 256, 0, stream>>>(xn, Wtv, bv, vb);

  attention_kernel<<<dim3(3, 3, 192), 256, 0, stream>>>(qb, kb, vb, dist, rts, attnO);

  gemm_bias_kernel<false><<<ggrid, 256, 0, stream>>>(attnO, Wto, bo, out);
}
